// SIES_GNN_59184649338988
// MI455X (gfx1250) — matrix operand order repaired and checked by reading, not compiled, not run on hardware
//
#include <hip/hip_runtime.h>

#define NFEAT  128
#define NHID   64
#define NCLASS 16
#define HEADS  4
#define HC     (HEADS * NHID)   // 256
#define DT_C   0.01f
#define EPS_C  1e-5f

typedef __attribute__((ext_vector_type(16))) __bf16 v16bf;
typedef __attribute__((ext_vector_type(8)))  __bf16 v8bf;
typedef __attribute__((ext_vector_type(4)))  __bf16 v4bf;
typedef __attribute__((ext_vector_type(8)))  float  v8f;

__device__ __forceinline__ __bf16 f2bf(float f) {
    unsigned u = __builtin_bit_cast(unsigned, f);
    u += 0x7FFFu + ((u >> 16) & 1u);            // round-to-nearest-even
    unsigned short h = (unsigned short)(u >> 16);
    return __builtin_bit_cast(__bf16, h);
}

// ---------------------------------------------------------------------------
// f32 -> bf16 row-major conversion, 4 elements / thread (b128 in, b64 out)
// ---------------------------------------------------------------------------
__global__ __launch_bounds__(256)
void cvt_bf16_kernel(const float* __restrict__ src, __bf16* __restrict__ dst, int n4) {
    int t = blockIdx.x * 256 + threadIdx.x;
    if (t >= n4) return;
    float4 v = ((const float4*)src)[t];
    v4bf o;
    o[0] = f2bf(v.x); o[1] = f2bf(v.y); o[2] = f2bf(v.z); o[3] = f2bf(v.w);
    ((v4bf*)dst)[t] = o;
}

// ---------------------------------------------------------------------------
// Pack W[K,Nc] (f32, row-major) into per-lane B-fragment layout (bf16):
//   WP[((tn*(K/32)+kb)*32 + lane)*16 + j] = W[kb*32 + 8*(lane>>4) + (j&7) + 16*(j>>3)][tn*16 + (lane&15)]
// matching the ISA 16-bit 32x16 B layout (lanes 0-15: K=0..7,16..23, lanes 16-31: K=8..15,24..31).
// ---------------------------------------------------------------------------
__global__ __launch_bounds__(256)
void pack_w_kernel(const float* __restrict__ W, __bf16* __restrict__ WP, int K, int Nc) {
    int t = blockIdx.x * 256 + threadIdx.x;
    if (t >= K * Nc) return;
    int j    = t & 15;
    int lane = (t >> 4) & 31;
    int rest = t >> 9;
    int kb   = rest % (K >> 5);
    int tn   = rest / (K >> 5);
    int col  = tn * 16 + (lane & 15);
    int k    = kb * 32 + (lane >> 4) * 8 + (j & 7) + (j >> 3) * 16;
    WP[t] = f2bf(W[(size_t)k * Nc + col]);
}

// ---------------------------------------------------------------------------
// GEMM: C[M,Nc] = act(A[M,K] @ W[K,Nc] + bias), A bf16 row-major, W pre-packed.
// One wave per 16x16 tile; inner loop = 2x b128 A loads + 2x b128 B loads + WMMA.
// Requires M%16==0, K%32==0, Nc%16==0 (true for all GEMMs here).
// ---------------------------------------------------------------------------
__global__ __launch_bounds__(256)
void gemm_wmma_bf16(const __bf16* __restrict__ A, const __bf16* __restrict__ WP,
                    const float* __restrict__ bias, float* __restrict__ C,
                    int M, int K, int Nc, int relu, int tiles) {
    int wave = blockIdx.x * 8 + (threadIdx.x >> 5);
    if (wave >= tiles) return;                    // wave-uniform, EXEC stays full
    int lane = threadIdx.x & 31;
    int ntiles = Nc >> 4;
    int tm = wave / ntiles;
    int tn = wave - tm * ntiles;
    int half = lane >> 4;
    int lr   = lane & 15;

    int arow = tm * 16 + lr;                      // A: lane holds row M=lr (both halves)
    int bcol = tn * 16 + lr;                      // B: lane holds col N=lr (both halves)

    const __bf16* ap = A  + (size_t)arow * K + half * 8;      // elems 0..7 ; +16 -> 8..15
    const __bf16* bp = WP + ((size_t)tn * (K >> 5) * 32 + lane) * 16;

    v8f acc = {};
    for (int k0 = 0; k0 < K; k0 += 32) {
        v8bf a0 = *(const v8bf*)ap;               // K = k0 + 8*half .. +7
        v8bf a1 = *(const v8bf*)(ap + 16);        // K = k0 + 16 + 8*half .. +7
        v16bf af = __builtin_shufflevector(a0, a1,
                       0, 1, 2, 3, 4, 5, 6, 7, 8, 9, 10, 11, 12, 13, 14, 15);
        v16bf bfv = *(const v16bf*)bp;            // contiguous pre-packed fragment
        acc = __builtin_amdgcn_wmma_f32_16x16x32_bf16(
                  false, af, false, bfv, (short)0, acc, false, false);
        ap += 32;
        bp += 512;                                // 32 lanes * 16 elems
    }

    float b = bias ? bias[bcol] : 0.0f;
    #pragma unroll
    for (int i = 0; i < 8; ++i) {
        int r = tm * 16 + half * 8 + i;           // C: VGPR i -> M = 8*half + i
        float v = acc[i] + b;
        if (relu) v = v > 0.0f ? v : 0.0f;
        C[(size_t)r * Nc + bcol] = v;
    }
}

// ---------------------------------------------------------------------------
// Edge / segment kernels
// ---------------------------------------------------------------------------
__device__ __forceinline__ void atomicMaxF(float* addr, float v) {
    if (v >= 0.0f) atomicMax((int*)addr, __float_as_int(v));
    else           atomicMin((unsigned int*)addr, __float_as_uint(v));
}

__global__ __launch_bounds__(256)
void deg_kernel(const int* __restrict__ col, float* __restrict__ deg, int E) {
    int t = blockIdx.x * 256 + threadIdx.x;
    if (t < E) atomicAdd(deg + col[t], 1.0f);
}

__global__ __launch_bounds__(256)
void dis_kernel(const float* __restrict__ deg, float* __restrict__ dis, int N) {
    int t = blockIdx.x * 256 + threadIdx.x;
    if (t < N) { float d = deg[t]; dis[t] = d > 0.0f ? 1.0f / sqrtf(d) : 0.0f; }
}

__global__ __launch_bounds__(256)
void norm_kernel(const int* __restrict__ row, const int* __restrict__ col,
                 const float* __restrict__ dis, float* __restrict__ nrm, int E) {
    int t = blockIdx.x * 256 + threadIdx.x;
    if (t < E) nrm[t] = dis[row[t]] * dis[col[t]];
}

// score[e,h] = sum_c leaky_relu(xl[row,h,c] + xr[col,h,c], 0.2) * att[h,c]
__global__ __launch_bounds__(256)
void score_kernel(const float* __restrict__ xl, const float* __restrict__ xr,
                  const float* __restrict__ att,
                  const int* __restrict__ row, const int* __restrict__ col,
                  float* __restrict__ score, int E) {
    int t = blockIdx.x * 256 + threadIdx.x;
    if (t >= E * HEADS) return;
    int e = t >> 2, h = t & 3;
    const float* lp = xl + (size_t)row[e] * HC + h * NHID;
    const float* rp = xr + (size_t)col[e] * HC + h * NHID;
    const float* ap = att + h * NHID;
    float s = 0.0f;
    #pragma unroll 8
    for (int i = 0; i < NHID; ++i) {
        float f = lp[i] + rp[i];
        f = f > 0.0f ? f : 0.2f * f;
        s += f * ap[i];
    }
    score[t] = s;
}

__global__ __launch_bounds__(256)
void smax_kernel(const float* __restrict__ score, const int* __restrict__ col,
                 float* __restrict__ smax, int E) {
    int t = blockIdx.x * 256 + threadIdx.x;
    if (t >= E * HEADS) return;
    int e = t >> 2, h = t & 3;
    atomicMaxF(smax + col[e] * HEADS + h, score[t]);
}

__global__ __launch_bounds__(256)
void expsum_kernel(const float* __restrict__ score, const float* __restrict__ smax,
                   const int* __restrict__ col, float* __restrict__ aexp,
                   float* __restrict__ ssum, int E) {
    int t = blockIdx.x * 256 + threadIdx.x;
    if (t >= E * HEADS) return;
    int e = t >> 2, h = t & 3;
    float a = expf(score[t] - smax[col[e] * HEADS + h]);
    aexp[t] = a;
    atomicAdd(ssum + col[e] * HEADS + h, a);
}

__global__ __launch_bounds__(256)
void wedge_kernel(const float* __restrict__ aexp, const float* __restrict__ ssum,
                  const float* __restrict__ nrm, const int* __restrict__ col,
                  float* __restrict__ wedge, int E) {
    int t = blockIdx.x * 256 + threadIdx.x;
    if (t >= E * HEADS) return;
    int e = t >> 2, h = t & 3;
    float s = ssum[col[e] * HEADS + h];
    s = s > 0.0f ? s : 1.0f;
    wedge[t] = aexp[t] / s * nrm[e];
}

// coupling[col, :] += v[row, :] * w_edge[e, head], 4 floats per thread
__global__ __launch_bounds__(256)
void scatter_kernel(const float* __restrict__ v, const float* __restrict__ wedge,
                    const int* __restrict__ row, const int* __restrict__ col,
                    float* __restrict__ coup, int E) {
    int t = blockIdx.x * 256 + threadIdx.x;
    if (t >= E * 64) return;
    int e = t >> 6, j = t & 63;                   // j = float4 index within 256-vector
    float4 vv = *(const float4*)(v + (size_t)row[e] * HC + 4 * j);
    float  w  = wedge[e * HEADS + (j >> 4)];      // head = (4j)/64
    float* dst = coup + (size_t)col[e] * HC + 4 * j;
    atomicAdd(dst + 0, vv.x * w);
    atomicAdd(dst + 1, vv.y * w);
    atomicAdd(dst + 2, vv.z * w);
    atomicAdd(dst + 3, vv.w * w);
}

// Damped-oscillator update + RMSNorm of X and Y. One wave per node row.
__global__ __launch_bounds__(256)
void update_kernel(float* __restrict__ Y, float* __restrict__ X,
                   const float* __restrict__ CY, int N,
                   float ay, float ax, float ac) {
    int node = blockIdx.x * 8 + (threadIdx.x >> 5);
    if (node >= N) return;
    int lane = threadIdx.x & 31;
    float* yp = Y + (size_t)node * NHID;
    float* xp = X + (size_t)node * NHID;
    const float* cp = CY + (size_t)node * NHID;

    float y0 = yp[lane], y1 = yp[lane + 32];
    float x0 = xp[lane], x1 = xp[lane + 32];
    float c0 = cp[lane], c1 = cp[lane + 32];

    float ny0 = ay * y0 + ax * x0 + ac * c0;
    float ny1 = ay * y1 + ax * x1 + ac * c1;
    float nx0 = x0 + DT_C * ny0;
    float nx1 = x1 + DT_C * ny1;

    float sy = ny0 * ny0 + ny1 * ny1;
    float sx = nx0 * nx0 + nx1 * nx1;
    #pragma unroll
    for (int m = 16; m >= 1; m >>= 1) {
        sy += __shfl_xor(sy, m, 32);
        sx += __shfl_xor(sx, m, 32);
    }
    float iy = 1.0f / sqrtf(sy * (1.0f / NHID) + EPS_C);
    float ix = 1.0f / sqrtf(sx * (1.0f / NHID) + EPS_C);

    yp[lane] = ny0 * iy; yp[lane + 32] = ny1 * iy;
    xp[lane] = nx0 * ix; xp[lane + 32] = nx1 * ix;
}

// ---------------------------------------------------------------------------
// Host orchestration
// ---------------------------------------------------------------------------
static inline void launch_gemm(const __bf16* A, const __bf16* WP, const float* b,
                               float* C, int M, int K, int Nc, int relu,
                               hipStream_t stream) {
    int tiles = (M / 16) * (Nc / 16);
    int blocks = (tiles + 7) / 8;
    gemm_wmma_bf16<<<blocks, 256, 0, stream>>>(A, WP, b, C, M, K, Nc, relu, tiles);
}

static inline void launch_cvt(const float* src, __bf16* dst, size_t n,
                              hipStream_t stream) {
    int n4 = (int)(n / 4);
    cvt_bf16_kernel<<<(n4 + 255) / 256, 256, 0, stream>>>(src, dst, n4);
}

static inline void launch_pack(const float* W, __bf16* WP, int K, int Nc,
                               hipStream_t stream) {
    int n = K * Nc;
    pack_w_kernel<<<(n + 255) / 256, 256, 0, stream>>>(W, WP, K, Nc);
}

extern "C" void kernel_launch(void* const* d_in, const int* in_sizes, int n_in,
                              void* d_out, int out_size, void* d_ws, size_t ws_size,
                              hipStream_t stream) {
    const float* x     = (const float*)d_in[0];
    const float* enc_w = (const float*)d_in[1];
    const float* enc_b = (const float*)d_in[2];
    const float* dec_w = (const float*)d_in[3];
    const float* dec_b = (const float*)d_in[4];
    const float* lo_w  = (const float*)d_in[5];
    const float* lo_b  = (const float*)d_in[6];
    const float* dpl_w = (const float*)d_in[7];
    const float* dpl_b = (const float*)d_in[8];
    const float* dpr_w = (const float*)d_in[9];
    const float* dpr_b = (const float*)d_in[10];
    const float* lin_w = (const float*)d_in[11];
    const float* lin_b = (const float*)d_in[12];
    const float* att   = (const float*)d_in[13];
    const int*   ei    = (const int*)d_in[14];

    const int N = in_sizes[0] / NFEAT;       // 10000
    const int E = in_sizes[14] / 2;          // 160000
    const int* erow = ei;
    const int* ecol = ei + E;

    // workspace carving (float units, 256B-aligned blocks)
    float* ws = (float*)d_ws;
    size_t off = 0;
    auto carve = [&](size_t nfloats) {
        float* p = ws + off; off += (nfloats + 63) & ~(size_t)63; return p;
    };
    float* Y     = carve((size_t)N * NHID);
    float* X     = carve((size_t)N * NHID);
    float* XL    = carve((size_t)N * HC);
    float* XR    = carve((size_t)N * HC);
    float* CY    = carve((size_t)N * NHID);
    float* DEG   = carve(N);
    float* DIS   = carve(N);
    float* NRM   = carve(E);
    float* SCORE = carve((size_t)E * HEADS);
    float* AEXP  = carve((size_t)E * HEADS);
    float* WEDGE = carve((size_t)E * HEADS);
    float* SMAX  = carve((size_t)N * HEADS);
    float* SSUM  = carve((size_t)N * HEADS);
    // bf16 scratch (sizes in bf16 elements -> /2 floats)
    __bf16* ABF1 = (__bf16*)carve((size_t)N * HC / 2);     // x / COUP in bf16
    __bf16* ABF2 = (__bf16*)carve((size_t)N * NHID / 2);   // Y / X in bf16
    __bf16* ENCP = (__bf16*)carve((size_t)NFEAT * NHID / 2);
    __bf16* DPLP = (__bf16*)carve((size_t)NHID * HC / 2);
    __bf16* DPRP = (__bf16*)carve((size_t)NHID * HC / 2);
    __bf16* LINP = (__bf16*)carve((size_t)NHID * HC / 2);
    __bf16* LOP  = (__bf16*)carve((size_t)HC * NHID / 2);
    __bf16* DECP = (__bf16*)carve((size_t)NHID * NCLASS / 2);
    float* V     = XL;   // attention projections dead after score phase
    float* COUP  = XR;

    // dynamics constants: softplus(1) = ln(1+e)
    const float sp = logf(1.0f + expf(1.0f));
    const float ay = 1.0f - DT_C * 2.0f * sp * sp;   // 1 - dt*2*zeta*omega
    const float ax = -DT_C * sp * sp;                // -dt*omega^2
    const float ac =  DT_C * sp;                     //  dt*Ks

    const int eb   = (E + 255) / 256;
    const int ehb  = (E * HEADS + 255) / 256;
    const int nb   = (N + 255) / 256;
    const int scb  = (E * 64 + 255) / 256;
    const int updb = (N + 7) / 8;

    // 0) pack all weights to bf16 fragment layout (once)
    launch_pack(enc_w, ENCP, NFEAT, NHID, stream);
    launch_pack(dpl_w, DPLP, NHID, HC, stream);
    launch_pack(dpr_w, DPRP, NHID, HC, stream);
    launch_pack(lin_w, LINP, NHID, HC, stream);
    launch_pack(lo_w,  LOP,  HC, NHID, stream);
    launch_pack(dec_w, DECP, NHID, NCLASS, stream);

    // 1) encoder + fixed attention projections
    launch_cvt(x, ABF1, (size_t)N * NFEAT, stream);
    launch_gemm(ABF1, ENCP, enc_b, Y, N, NFEAT, NHID, 1, stream);
    launch_cvt(Y, ABF2, (size_t)N * NHID, stream);
    launch_gemm(ABF2, DPLP, dpl_b, XL, N, NHID, HC, 0, stream);
    launch_gemm(ABF2, DPRP, dpr_b, XR, N, NHID, HC, 0, stream);

    // 2) degrees + GCN norm
    hipMemsetAsync(DEG, 0, (size_t)N * sizeof(float), stream);
    deg_kernel<<<eb, 256, 0, stream>>>(ecol, DEG, E);
    dis_kernel<<<nb, 256, 0, stream>>>(DEG, DIS, N);
    norm_kernel<<<eb, 256, 0, stream>>>(erow, ecol, DIS, NRM, E);

    // 3) fixed edge attention -> w_edge
    score_kernel<<<ehb, 256, 0, stream>>>(XL, XR, att, erow, ecol, SCORE, E);
    hipMemsetAsync(SMAX, 0xFF, (size_t)N * HEADS * sizeof(float), stream); // -NaN sentinel
    smax_kernel<<<ehb, 256, 0, stream>>>(SCORE, ecol, SMAX, E);
    hipMemsetAsync(SSUM, 0, (size_t)N * HEADS * sizeof(float), stream);
    expsum_kernel<<<ehb, 256, 0, stream>>>(SCORE, SMAX, ecol, AEXP, SSUM, E);
    wedge_kernel<<<ehb, 256, 0, stream>>>(AEXP, SSUM, NRM, ecol, WEDGE, E);

    // 4) X = Y
    hipMemcpyAsync(X, Y, (size_t)N * NHID * sizeof(float),
                   hipMemcpyDeviceToDevice, stream);

    // 5) layers
    for (int l = 0; l < 4; ++l) {
        launch_cvt(Y, ABF2, (size_t)N * NHID, stream);
        launch_gemm(ABF2, LINP, lin_b, V, N, NHID, HC, 0, stream);
        hipMemsetAsync(COUP, 0, (size_t)N * HC * sizeof(float), stream);
        scatter_kernel<<<scb, 256, 0, stream>>>(V, WEDGE, erow, ecol, COUP, E);
        launch_cvt(COUP, ABF1, (size_t)N * HC, stream);
        launch_gemm(ABF1, LOP, lo_b, CY, N, HC, NHID, 0, stream);
        update_kernel<<<updb, 256, 0, stream>>>(Y, X, CY, N, ay, ax, ac);
    }

    // 6) decoder
    launch_cvt(X, ABF2, (size_t)N * NHID, stream);
    launch_gemm(ABF2, DECP, dec_b, (float*)d_out, N, NHID, NCLASS, 0, stream);
}
